// MemoryBridge_74895639708119
// MI455X (gfx1250) — compile-verified, matrix operand order checked by reading, not run on hardware
//
#include <hip/hip_runtime.h>
#include <hip/hip_bf16.h>

// ---------------------------------------------------------------------------
// MemoryBridge on MI455X (gfx1250): fused flash-style memory hops.
//   - v_wmma_f32_16x16x32_bf16 for QK^T and PV GEMMs
//   - double-buffered GLOBAL_LOAD_ASYNC_TO_LDS_B128 tile staging (ASYNCcnt)
//   - all LDS traffic through explicit addrspace(3) pointers -> ds_load_b128
// ---------------------------------------------------------------------------

typedef __attribute__((ext_vector_type(16))) __bf16 v16bf;
typedef __attribute__((ext_vector_type(8)))  __bf16 v8bf;
typedef __attribute__((ext_vector_type(8)))  float  v8f;

constexpr int BN     = 2048;
constexpr int KWD    = 1000;
constexpr int EMB    = 300;
constexpr int EMBP   = 320;            // padded to 10 x 32 (bf16 WMMA K)
constexpr int VOCAB  = 50000;
constexpr int VOCABP = 50016;          // padded to 1563 * 32
constexpr int EDIM   = 512;
constexpr int KSTEPS = EMBP / 32;      // 10 k-steps per QK tile
constexpr int NTILES = EMBP / 16;      // 20 n-tiles for the PV accumulator
constexpr int ROWS_WG  = 128;          // rows per workgroup (8 waves x 16)
constexpr int NSPLIT   = 16;           // vocab splits (softmax partials)
constexpr int VCHUNK   = 3136;         // 98*32, 16*3136 >= 50000

constexpr int TILE_E = 32 * EMBP;      // elements per staged K or V tile (20KB)
// LDS: Q | K0 | K1 | V0 | V1 | P
constexpr int LDS_Q = ROWS_WG * EMBP;
constexpr int LDS_P = 8 * 16 * 32;
constexpr size_t HOP_LDS_BYTES = (size_t)(LDS_Q + 4 * TILE_E + LDS_P) * 2;

// ---- CDNA5 async-copy support (guarded; falls back to sync copies) --------
#if __has_builtin(__builtin_amdgcn_global_load_async_to_lds_b128) && \
    __has_builtin(__builtin_amdgcn_s_wait_asynccnt)
#define USE_ASYNC 1
#else
#define USE_ASYNC 0
#endif

// async builtin expects int4-vector pointers: AS1 global side, AS3 LDS side
typedef int v4i __attribute__((vector_size(16)));
typedef __attribute__((address_space(1))) v4i* gptr_t;
typedef __attribute__((address_space(3))) v4i* lptr_t;

// explicit LDS (addrspace(3)) element/vector types so every LDS access
// provably lowers to ds_load/ds_store (no FLAT fallback on pointer selects)
typedef __attribute__((address_space(3))) __bf16 lbf16;
typedef __attribute__((address_space(3))) v8bf   lv8bf;

__device__ __forceinline__ gptr_t gcast(const void* p) {
  // global generic pointer bits == addrspace(1) bits
  return (gptr_t)(unsigned long long)(uintptr_t)p;
}
__device__ __forceinline__ lbf16* lds_base(void* p) {
  // low 32 bits of a generic LDS pointer are the LDS byte offset
  return (lbf16*)(unsigned)(uintptr_t)p;
}

#if USE_ASYNC
#define ASYNC_WAIT(n) __builtin_amdgcn_s_wait_asynccnt(n)
#else
#define ASYNC_WAIT(n) ((void)0)
#endif

__device__ __forceinline__ v16bf load16l(const lbf16* p) {
  // elements 0..7 = k..k+7, elements 8..15 = k+16..k+23 (A/B 16-bit layouts)
  union { v16bf v; v8bf h[2]; } u;
  u.h[0] = *(const lv8bf*)p;
  u.h[1] = *(const lv8bf*)(p + 16);
  return u.v;
}

// stage one K tile + one V tile (each 32x320 bf16, contiguous in global & LDS)
__device__ __forceinline__ void stage_pair(const __bf16* gk, const __bf16* gv,
                                           lbf16* lk, lbf16* lv, int tid) {
  constexpr int CH = TILE_E / 8;          // 1280 x 16B chunks per tile
#if USE_ASYNC
  for (int c = tid; c < CH; c += 256) {   // 5+5 async instructions per wave
    __builtin_amdgcn_global_load_async_to_lds_b128(gcast(gk + c * 8),
                                                   (lptr_t)(lk + c * 8), 0, 0);
    __builtin_amdgcn_global_load_async_to_lds_b128(gcast(gv + c * 8),
                                                   (lptr_t)(lv + c * 8), 0, 0);
  }
#else
  for (int c = tid; c < CH; c += 256) {
    *(lv8bf*)(lk + c * 8) = *(const v8bf*)(gk + c * 8);
    *(lv8bf*)(lv + c * 8) = *(const v8bf*)(gv + c * 8);
  }
#endif
}

// ------------------------- input MLP: feature = (logits*mask)@W^T + b -------
__global__ __launch_bounds__(256) void mb_mlp(const float* __restrict__ logits,
                                              const float* __restrict__ mask,
                                              const float* __restrict__ w,
                                              const float* __restrict__ bias,
                                              float* __restrict__ feat,
                                              __bf16* __restrict__ featbf) {
  int idx = blockIdx.x * blockDim.x + threadIdx.x;
  if (idx >= BN * EMBP) return;
  int b = idx / EMBP, e = idx % EMBP;
  float s = 0.f;
  if (e < EMB) {
    const float* lr = logits + (size_t)b * KWD;
    const float* mr = mask   + (size_t)b * KWD;
    const float* wr = w      + (size_t)e * KWD;
    for (int k = 0; k < KWD; ++k) s += lr[k] * mr[k] * wr[k];
    s += bias[e];
  }
  feat[idx]   = (e < EMB) ? s : 0.f;
  featbf[idx] = (__bf16)((e < EMB) ? s : 0.f);
}

// ------- embedding -> bf16, 32-row tile-transposed: embT[jt][col][kr] -------
__global__ __launch_bounds__(256) void mb_conv_embT(const float* __restrict__ emb,
                                                    __bf16* __restrict__ embT) {
  size_t total  = (size_t)VOCABP * EMBP;
  size_t stride = (size_t)gridDim.x * blockDim.x;
  for (size_t idx = (size_t)blockIdx.x * blockDim.x + threadIdx.x;
       idx < total; idx += stride) {
    size_t jt = idx / (EMBP * 32);
    int rem = (int)(idx % (EMBP * 32));
    int e = rem >> 5, kr = rem & 31;
    size_t v = jt * 32 + kr;
    embT[idx] = (__bf16)((v < VOCAB && e < EMB) ? emb[v * EMB + e] : 0.f);
  }
}

// ---- mem_keys [EMB][VOCAB] -> bf16 kT [VOCABP][EMBP] (zero-padded) ---------
__global__ __launch_bounds__(256) void mb_conv_keys(const float* __restrict__ mk,
                                                    __bf16* __restrict__ kT) {
  __shared__ float tile[32][33];
  int v0 = blockIdx.x * 32;
  int k0 = blockIdx.y * 32;
  int tx = threadIdx.x & 31, ty = threadIdx.x >> 5;   // 32 x 8
  for (int i = 0; i < 32; i += 8) {
    int k = k0 + ty + i, v = v0 + tx;
    tile[ty + i][tx] = (k < EMB && v < VOCAB) ? mk[(size_t)k * VOCAB + v] : 0.f;
  }
  __syncthreads();
  for (int i = 0; i < 32; i += 8) {
    int v = v0 + ty + i, k = k0 + tx;                 // v < VOCABP always
    kT[(size_t)v * EMBP + k] = (__bf16)tile[tx][ty + i];
  }
}

// ------------------------- fused hop: flash attention partials --------------
__global__ __launch_bounds__(256) void mb_hop(const __bf16* __restrict__ qbf,
                                              const __bf16* __restrict__ kT,
                                              const __bf16* __restrict__ vT,
                                              float* __restrict__ opart,
                                              float* __restrict__ mpart,
                                              float* __restrict__ lpart) {
  extern __shared__ char smem[];
  lbf16* LQ = lds_base(smem);                 // [row 0..127][k 0..319]
  lbf16* LK0 = LQ + LDS_Q;                    // K buffers, [col][k]
  lbf16* LV0 = LQ + LDS_Q + 2 * TILE_E;       // V buffers, [col][k 0..31]
  lbf16* LP  = LQ + LDS_Q + 4 * TILE_E;       // per-wave P [m][k 0..31]

  const int tid  = threadIdx.x;
  const int wave = tid >> 5;
  const int lane = tid & 31;
  const int l15  = lane & 15;
  const int hh   = lane >> 4;              // which K-half of the 16-bit layouts
  const int row0 = blockIdx.x * ROWS_WG;
  const int split = blockIdx.y;
  const int v0   = split * VCHUNK;
  const int vend = (v0 + VCHUNK < VOCAB) ? (v0 + VCHUNK) : VOCAB;
  const int myrow = wave * 16;

  // stage Q once: 128x320 bf16 (20 async b128 per wave)
#if USE_ASYNC
  for (int c = tid; c < LDS_Q / 8; c += 256)
    __builtin_amdgcn_global_load_async_to_lds_b128(
        gcast(qbf + (size_t)row0 * EMBP + c * 8), (lptr_t)(LQ + c * 8), 0, 0);
#else
  for (int c = tid; c < LDS_Q / 8; c += 256)
    *(lv8bf*)(LQ + c * 8) = *(const v8bf*)(qbf + (size_t)row0 * EMBP + c * 8);
#endif

  v8f acc[NTILES];
#pragma unroll
  for (int t = 0; t < NTILES; ++t) acc[t] = (v8f){};
  float mrow[8], lrow[8];
#pragma unroll
  for (int r = 0; r < 8; ++r) { mrow[r] = -3.0e38f; lrow[r] = 0.f; }

  // prologue: stage first tile pair into buffer 0
  stage_pair(kT + (size_t)v0 * EMBP, vT + (size_t)(v0 >> 5) * TILE_E,
             LK0, LV0, tid);

  int idx = 0;
  for (int j = v0; j < vend; j += 32, ++idx) {
    const int cur = idx & 1;
    const bool hn = (j + 32) < vend;
    if (hn)                                   // prefetch next tile pair
      stage_pair(kT + (size_t)(j + 32) * EMBP,
                 vT + (size_t)((j + 32) >> 5) * TILE_E,
                 LK0 + (cur ^ 1) * TILE_E, LV0 + (cur ^ 1) * TILE_E, tid);
    if (hn) ASYNC_WAIT(10); else ASYNC_WAIT(0);   // older batch complete
    __syncthreads();                              // visible to all waves
    lbf16* ldsK = LK0 + cur * TILE_E;             // AS3 offset arithmetic only
    lbf16* ldsV = LV0 + cur * TILE_E;

    // ---- S = Q @ K-tile : two 16x16 f32 tiles, 10 k-steps each
    v8f s0 = (v8f){}, s1 = (v8f){};
#pragma unroll
    for (int ks = 0; ks < KSTEPS; ++ks) {
      const int kb = ks * 32 + hh * 8;
      v16bf aop = load16l(LQ + (myrow + l15) * EMBP + kb);
      v16bf b0  = load16l(ldsK + l15 * EMBP + kb);
      v16bf b1  = load16l(ldsK + (16 + l15) * EMBP + kb);
      s0 = __builtin_amdgcn_wmma_f32_16x16x32_bf16(false, aop, false, b0,
                                                   (short)0, s0, false, false);
      s1 = __builtin_amdgcn_wmma_f32_16x16x32_bf16(false, aop, false, b1,
                                                   (short)0, s1, false, false);
    }

    // ---- online softmax over these 32 columns
    const bool ok0 = (j + l15) < vend;
    const bool ok1 = (j + 16 + l15) < vend;
    v8f alphav = (v8f){};
#pragma unroll
    for (int r = 0; r < 8; ++r) {          // rows m = r + 8*hh
      float a0 = ok0 ? s0[r] : -3.0e38f;
      float a1 = ok1 ? s1[r] : -3.0e38f;
      float mx = fmaxf(a0, a1);
#pragma unroll
      for (int off = 1; off < 16; off <<= 1) mx = fmaxf(mx, __shfl_xor(mx, off, 32));
      float mnew  = fmaxf(mrow[r], mx);
      float alpha = __expf(mrow[r] - mnew);
      float p0 = ok0 ? __expf(a0 - mnew) : 0.f;
      float p1 = ok1 ? __expf(a1 - mnew) : 0.f;
      float ps = p0 + p1;
#pragma unroll
      for (int off = 1; off < 16; off <<= 1) ps += __shfl_xor(ps, off, 32);
      lrow[r] = lrow[r] * alpha + ps;
      mrow[r] = mnew;
      alphav[r] = alpha;
      // P in C-layout -> wave-private LDS scratch (ds_store_b16)
      LP[wave * 512 + (r + 8 * hh) * 32 + l15]      = (__bf16)p0;
      LP[wave * 512 + (r + 8 * hh) * 32 + 16 + l15] = (__bf16)p1;
    }
#pragma unroll
    for (int t = 0; t < NTILES; ++t) acc[t] *= alphav;

    __builtin_amdgcn_wave_barrier();       // keep P stores before P loads

    // ---- O += P @ V-tile (K = 32 vocab rows)
    v16bf pa = load16l(LP + wave * 512 + l15 * 32 + hh * 8);
#pragma unroll
    for (int t = 0; t < NTILES; ++t) {
      v16bf bv = load16l(ldsV + (t * 16 + l15) * 32 + hh * 8);
      acc[t] = __builtin_amdgcn_wmma_f32_16x16x32_bf16(false, pa, false, bv,
                                                       (short)0, acc[t], false, false);
    }
    __syncthreads();                       // all waves done reading this buffer
  }

  // ---- write partial O / m / l for this vocab split
#pragma unroll
  for (int t = 0; t < NTILES; ++t) {
#pragma unroll
    for (int r = 0; r < 8; ++r) {
      int grow = row0 + myrow + r + 8 * hh;
      opart[((size_t)split * BN + grow) * EMBP + t * 16 + l15] = acc[t][r];
    }
  }
  if (l15 == 0) {
#pragma unroll
    for (int r = 0; r < 8; ++r) {
      int grow = row0 + myrow + r + 8 * hh;
      mpart[(size_t)split * BN + grow] = mrow[r];
      lpart[(size_t)split * BN + grow] = lrow[r];
    }
  }
}

// ------------------------- combine partial softmax + feature update ---------
__global__ __launch_bounds__(64) void mb_combine(const float* __restrict__ opart,
                                                 const float* __restrict__ mpart,
                                                 const float* __restrict__ lpart,
                                                 float* __restrict__ feat,
                                                 __bf16* __restrict__ featbf) {
  int row = blockIdx.x;
  float mstar = -3.0e38f;
  for (int s = 0; s < NSPLIT; ++s) mstar = fmaxf(mstar, mpart[(size_t)s * BN + row]);
  float w[NSPLIT];
  float denom = 0.f;
  for (int s = 0; s < NSPLIT; ++s) {
    w[s] = __expf(mpart[(size_t)s * BN + row] - mstar);
    denom += w[s] * lpart[(size_t)s * BN + row];
  }
  float inv = 1.f / denom;
  for (int c = threadIdx.x; c < EMBP; c += blockDim.x) {
    float a = 0.f;
    for (int s = 0; s < NSPLIT; ++s)
      a += w[s] * opart[((size_t)s * BN + row) * EMBP + c];
    float nf = (c < EMB) ? feat[(size_t)row * EMBP + c] + a * inv : 0.f;
    feat[(size_t)row * EMBP + c]   = nf;
    featbf[(size_t)row * EMBP + c] = (__bf16)nf;
  }
}

// ------------------------- enc/dec heads ------------------------------------
__global__ __launch_bounds__(256) void mb_heads(const float* __restrict__ feat,
                                                const float* __restrict__ encw,
                                                const float* __restrict__ encb,
                                                const float* __restrict__ decw,
                                                const float* __restrict__ decb,
                                                float* __restrict__ out) {
  int idx = blockIdx.x * blockDim.x + threadIdx.x;   // over 2*BN*EDIM
  if (idx >= 2 * BN * EDIM) return;
  int which = idx / (BN * EDIM);
  int r = idx % (BN * EDIM);
  int b = r / EDIM, jj = r % EDIM;
  const float* W  = which ? decw : encw;
  const float* bb = which ? decb : encb;
  const float* fr = feat + (size_t)b * EMBP;
  const float* wr = W + (size_t)jj * EMB;
  float s = bb[jj];
  for (int e = 0; e < EMB; ++e) s += fr[e] * wr[e];
  out[idx] = s;
}

// ---------------------------------------------------------------------------
extern "C" void kernel_launch(void* const* d_in, const int* in_sizes, int n_in,
                              void* d_out, int out_size, void* d_ws, size_t ws_size,
                              hipStream_t stream) {
  const float* logits  = (const float*)d_in[0];
  const float* kwdmask = (const float*)d_in[1];
  const float* embed   = (const float*)d_in[2];
  const float* mlp_w   = (const float*)d_in[3];
  const float* mlp_b   = (const float*)d_in[4];
  const float* mk0     = (const float*)d_in[5];
  const float* mk1     = (const float*)d_in[6];
  const float* enc_w   = (const float*)d_in[7];
  const float* enc_b   = (const float*)d_in[8];
  const float* dec_w   = (const float*)d_in[9];
  const float* dec_b   = (const float*)d_in[10];
  float* out = (float*)d_out;

  // workspace carve (256B aligned)
  char* ws = (char*)d_ws;
  auto align = [](size_t x) { return (x + 255) & ~(size_t)255; };
  size_t off = 0;
  float*  feat   = (float*)(ws + off);  off = align(off + (size_t)BN * EMBP * 4);
  __bf16* featbf = (__bf16*)(ws + off); off = align(off + (size_t)BN * EMBP * 2);
  __bf16* embT   = (__bf16*)(ws + off); off = align(off + (size_t)VOCABP * EMBP * 2);
  __bf16* kTbf   = (__bf16*)(ws + off); off = align(off + (size_t)VOCABP * EMBP * 2);
  float*  opart  = (float*)(ws + off);  off = align(off + (size_t)NSPLIT * BN * EMBP * 4);
  float*  mpart  = (float*)(ws + off);  off = align(off + (size_t)NSPLIT * BN * 4);
  float*  lpart  = (float*)(ws + off);  off = align(off + (size_t)NSPLIT * BN * 4);
  (void)off; (void)ws_size; (void)in_sizes; (void)n_in; (void)out_size;

  // 1) input MLP -> feature (f32 + bf16 padded)
  mb_mlp<<<(BN * EMBP + 255) / 256, 256, 0, stream>>>(logits, kwdmask, mlp_w, mlp_b,
                                                      feat, featbf);
  // 2) embedding -> bf16, 32-row tile-transposed V operand (both hops)
  mb_conv_embT<<<4096, 256, 0, stream>>>(embed, embT);

  // 3) two memory hops
  const float* mks[2] = {mk0, mk1};
  for (int h = 0; h < 2; ++h) {
    mb_conv_keys<<<dim3(VOCABP / 32, EMBP / 32), 256, 0, stream>>>(mks[h], kTbf);
    mb_hop<<<dim3(BN / ROWS_WG, NSPLIT), 256, HOP_LDS_BYTES, stream>>>(
        featbf, kTbf, embT, opart, mpart, lpart);
    mb_combine<<<BN, 64, 0, stream>>>(opart, mpart, lpart, feat, featbf);
  }

  // 4) enc/dec heads -> d_out (e_feature then d_feature)
  mb_heads<<<(2 * BN * EDIM + 255) / 256, 256, 0, stream>>>(feat, enc_w, enc_b,
                                                            dec_w, dec_b, out);
}
